// KANLayer_26688926777600
// MI455X (gfx1250) — matrix operand order repaired and checked by reading, not compiled, not run on hardware
//
#include <hip/hip_runtime.h>

// ---------------------------------------------------------------------------
// KAN layer on MI455X (gfx1250):
//   out[b,k] = bias[k] + sum_{i,j} B3(x[b,i]; knots[i])_j * W[k,i,j]
//   B=4096, F=512 (in), K=512 (out), num_basis=30 (padded to 32), order=3.
//
// The einsum is a [4096 x 15360] x [15360 x 512] GEMM = 64.4 GFLOP, far above
// the HBM roofline (~48 MB fundamental traffic -> ~2us at 23.3 TB/s), so the
// kernel must ride the WMMA pipe.  bf16 WMMA (v_wmma_f32_16x16x32_bf16) does
// K=32 per instruction = exactly one padded feature's basis block, 8x the MAC
// rate of the f32 16x16x4 path; basis values are in [0,1], weights ~N(0,1),
// so bf16 A/B with f32 accumulate is the right precision for this chip.
//
// The cubic basis is 4-sparse of 30, so it is evaluated on the fly (de Boor,
// using the real knot vectors) into a double-buffered LDS tile in bf16 rather
// than materializing a 250 MB dense basis tensor.  Weights are repacked once
// per launch into Wt[i][k][j0..31] bf16 (16.8 MB scratch, L2-resident) so a
// WMMA B-fragment is two 16B runs per lane.
//
// v2: all reciprocals in the basis path use v_rcp_f32 (__builtin_amdgcn_rcpf)
// instead of full IEEE division sequences — the results feed bf16 WMMA inputs
// so ~1 ulp is plenty, and this halves the serial VALU chain that must
// co-execute with the XDL pipe each iteration.
// ---------------------------------------------------------------------------

typedef __attribute__((ext_vector_type(16))) __bf16 v16bf;
typedef __attribute__((ext_vector_type(8)))  float  v8f;

namespace {
constexpr int Ff   = 512;   // in_features
constexpr int Kk   = 512;   // out_features
constexpr int NB   = 30;    // num_basis
constexpr int MK   = 34;    // knots per feature = NB + order + 1
constexpr int BM   = 64;    // batch rows per block
constexpr int BN   = 128;   // out cols per block
constexpr int ASTR = 40;    // bf16 elems per LDS A row (80B, padded vs 64B)
}

union FragAB { uint4 u[2]; v16bf v; };
union Pack32 { __bf16 h[32]; uint4 u[4]; };

__device__ __forceinline__ float fast_rcp(float x) {
  return __builtin_amdgcn_rcpf(x);   // v_rcp_f32, ~1 ulp; knots are distinct
}

// ---- prepass: W[k][i][0..29] f32  ->  Wt[i][k][0..31] bf16 (j padded) ------
__global__ __launch_bounds__(256) void kan_pack_weights(
    const float* __restrict__ W, __bf16* __restrict__ Wt) {
  const int t = blockIdx.x * 256 + threadIdx.x;   // 0 .. Kk*Ff-1
  const int k = t & (Kk - 1);                     // fastest -> coalesced store
  const int i = t >> 9;
  const float2* src = (const float2*)(W + ((size_t)k * Ff + i) * NB); // 8B aligned
  Pack32 p;
#pragma unroll
  for (int jj = 0; jj < 15; ++jj) {
    float2 v = src[jj];
    p.h[2 * jj]     = (__bf16)v.x;
    p.h[2 * jj + 1] = (__bf16)v.y;
  }
  p.h[30] = (__bf16)0.0f;
  p.h[31] = (__bf16)0.0f;
  uint4* dst = (uint4*)(Wt + (((size_t)i * Kk + k) << 5));
#pragma unroll
  for (int q = 0; q < 4; ++q) dst[q] = p.u[q];
}

// ---- fused basis + WMMA GEMM ----------------------------------------------
__global__ __launch_bounds__(256) void kan_wmma(
    const float* __restrict__ X, const float* __restrict__ knots,
    const __bf16* __restrict__ Wt, const float* __restrict__ bias,
    float* __restrict__ Out) {
  __shared__ alignas(16) __bf16 Abuf[2][BM][ASTR];   // double-buffered, 10 KB

  const int tid  = threadIdx.x;
  const int lane = tid & 31;
  const int wid  = tid >> 5;
  const int wm   = wid & 1;      // 2 wave rows (M)
  const int wn   = wid >> 1;     // 4 wave cols (N)
  const int lr   = lane & 15;
  const int hl   = lane >> 4;
  const int bm0  = blockIdx.x * BM;
  const int n0   = blockIdx.y * BN;

  // De Boor evaluation of the 4 nonzero cubic basis values for feature i,
  // scattered into a zeroed 32-slot bf16 row of the LDS A tile.
  auto compute_basis = [&](int i, int buf) {
    if (tid < BM) {
      const int m = tid;
      const float x = X[(size_t)(bm0 + m) * Ff + i];
      const float* kt = knots + (size_t)i * MK;      // setup knots are sorted
      const float kt0  = kt[0];
      const float invh = fast_rcp(kt[1] - kt0);      // uniform-grid locator
      int s = (int)floorf((x - kt0) * invh);
      s = s < 3 ? 3 : (s > NB - 1 ? NB - 1 : s);     // s in [3, 29]
      const float tm2 = kt[s - 2], tm1 = kt[s - 1], t0v = kt[s],
                  tp1 = kt[s + 1], tp2 = kt[s + 2], tp3 = kt[s + 3];
      float N0 = 1.0f, N1, N2, N3;
      { float w = fast_rcp(tp1 - t0v);
        N0 = (tp1 - x) * w;  N1 = (x - t0v) * w; }
      { float w0 = N0 * fast_rcp(tp1 - tm1);
        float n0v = (tp1 - x) * w0; float sv = (x - tm1) * w0;
        float w1 = N1 * fast_rcp(tp2 - t0v);
        N1 = sv + (tp2 - x) * w1;  N2 = (x - t0v) * w1;
        N0 = n0v; }
      { float w0 = N0 * fast_rcp(tp1 - tm2);
        float n0v = (tp1 - x) * w0; float sv = (x - tm2) * w0;
        float w1 = N1 * fast_rcp(tp2 - tm1);
        float n1v = sv + (tp2 - x) * w1; sv = (x - tm1) * w1;
        float w2 = N2 * fast_rcp(tp3 - t0v);
        N2 = sv + (tp3 - x) * w2;  N3 = (x - t0v) * w2;
        N0 = n0v; N1 = n1v; }
      const float valid =
          (x >= kt[3] && x <= kt[MK - 1 - 3]) ? 1.0f : 0.0f;  // base interval
      __bf16* row = &Abuf[buf][m][0];
      const uint4 z = {0u, 0u, 0u, 0u};
      *(uint4*)(row)      = z;
      *(uint4*)(row + 8)  = z;
      *(uint4*)(row + 16) = z;
      *(uint4*)(row + 24) = z;
      row[s - 3 + 0] = (__bf16)(N0 * valid);
      row[s - 3 + 1] = (__bf16)(N1 * valid);
      row[s - 3 + 2] = (__bf16)(N2 * valid);
      row[s - 3 + 3] = (__bf16)(N3 * valid);
    }
  };

  v8f acc[2][2] = {};          // 2x2 16x16 f32 tiles per wave (32 VGPRs)

  compute_basis(0, 0);
  __syncthreads();

  for (int i = 0; i < Ff; ++i) {
    const int cur = i & 1;
    FragAB a[2], b[2];

    // A fragments from LDS per the 16-bit 16x32 A layout:
    // lane<16: K0-7 & K16-23 of row lr; lane>=16: K8-15 & K24-31.
    #pragma unroll
    for (int t = 0; t < 2; ++t) {
      const __bf16* ap = &Abuf[cur][(wm * 2 + t) * 16 + lr][hl * 8];
      a[t].u[0] = *(const uint4*)(ap);
      a[t].u[1] = *(const uint4*)(ap + 16);
    }
    // B fragments straight from bf16-packed Wt (two 16B runs per lane):
    // lane<16: j0-7 & j16-23 of column kcol; lane>=16: j8-15 & j24-31.
    #pragma unroll
    for (int t = 0; t < 2; ++t) {
      const int kcol = n0 + (wn * 2 + t) * 16 + lr;
      const __bf16* bp = Wt + (((size_t)i * Kk + kcol) << 5) + hl * 8;
      b[t].u[0] = *(const uint4*)(bp);
      b[t].u[1] = *(const uint4*)(bp + 16);
      if (i + 1 < Ff) __builtin_prefetch(bp + ((size_t)Kk << 5), 0, 1);
    }

    // Overlap: evaluate next feature's basis into the other LDS buffer.
    if (i + 1 < Ff) compute_basis(i + 1, cur ^ 1);
    __syncthreads();

    acc[0][0] = __builtin_amdgcn_wmma_f32_16x16x32_bf16(
        false, a[0].v, false, b[0].v, (short)0, acc[0][0], false, false);
    acc[0][1] = __builtin_amdgcn_wmma_f32_16x16x32_bf16(
        false, a[0].v, false, b[1].v, (short)0, acc[0][1], false, false);
    acc[1][0] = __builtin_amdgcn_wmma_f32_16x16x32_bf16(
        false, a[1].v, false, b[0].v, (short)0, acc[1][0], false, false);
    acc[1][1] = __builtin_amdgcn_wmma_f32_16x16x32_bf16(
        false, a[1].v, false, b[1].v, (short)0, acc[1][1], false, false);
  }

  // Epilogue: C layout is VGPR r, lanes 0-15 -> M=r, lanes 16-31 -> M=r+8.
  #pragma unroll
  for (int mt = 0; mt < 2; ++mt)
    #pragma unroll
    for (int nt = 0; nt < 2; ++nt) {
      const int m0  = bm0 + (wm * 2 + mt) * 16 + hl * 8;
      const int col = n0 + (wn * 2 + nt) * 16 + lr;
      const float bc = bias[col];
      #pragma unroll
      for (int r = 0; r < 8; ++r)
        Out[(size_t)(m0 + r) * Kk + col] = acc[mt][nt][r] + bc;
    }
}

// ---------------------------------------------------------------------------
extern "C" void kernel_launch(void* const* d_in, const int* in_sizes, int n_in,
                              void* d_out, int out_size, void* d_ws, size_t ws_size,
                              hipStream_t stream) {
  const float* X     = (const float*)d_in[0];   // [4096, 512] f32
  const float* knots = (const float*)d_in[1];   // [512, 34]   f32 (sorted)
  const float* W     = (const float*)d_in[2];   // [512, 512, 30] f32
  const float* bias  = (const float*)d_in[3];   // [512] f32
  float* Out = (float*)d_out;                   // [4096, 512] f32
  __bf16* Wt = (__bf16*)d_ws;                   // 512*512*32*2 = 16.8 MB scratch

  const int batch = in_sizes[0] / Ff;           // 4096

  kan_pack_weights<<<(Kk * Ff) / 256, 256, 0, stream>>>(W, Wt);
  kan_wmma<<<dim3(batch / BM, Kk / BN), 256, 0, stream>>>(X, knots, Wt, bias, Out);
}
